// MultiHeadAttention_5488968204345
// MI455X (gfx1250) — matrix operand order repaired and checked by reading, not compiled, not run on hardware
//
#include <hip/hip_runtime.h>
#include <hip/hip_bf16.h>

typedef __attribute__((ext_vector_type(16))) _Float16 v16h;
typedef __attribute__((ext_vector_type(8)))  _Float16 v8h;
typedef __attribute__((ext_vector_type(4)))  _Float16 v4h;
typedef __attribute__((ext_vector_type(8)))  float    v8f;

constexpr int BSZ = 2;
constexpr int SEQ = 2048;
constexpr int DM  = 768;
constexpr int NH  = 12;
constexpr int DK  = 64;

#define DEV __device__ __forceinline__

DEV v8f wmma_f16(v16h a, v16h b, v8f c) {
  // D = A(16x32 f16) * B(32x16 f16) + C(16x16 f32)
  return __builtin_amdgcn_wmma_f32_16x16x32_f16(
      /*neg_a=*/false, a, /*neg_b=*/false, b,
      /*c_mod=*/(short)0, c, /*reuse_a=*/false, /*reuse_b=*/false);
}

DEV v16h cat8(v8h lo, v8h hi) {
  v16h r;
#pragma unroll
  for (int e = 0; e < 8; ++e) { r[e] = lo[e]; r[8 + e] = hi[e]; }
  return r;
}

// ---------------------------------------------------------------------------
// C = A @ W^T + bias   (torch Linear)
//   A: [M=4096, K=768]  fp32 (A_F32) or f16
//   W: [N=768,  K=768]  fp32
//   Out: OUT_HEAD -> f16 head-split [B, H, S, DK]; else fp32 row-major [M, N]
// Block: 256 threads (8 waves), tile 128x128, k-chunk 32.
// ---------------------------------------------------------------------------
template <bool A_F32, bool OUT_HEAD>
__global__ __launch_bounds__(256) void gemm_wt_kernel(
    const void* __restrict__ A_, const float* __restrict__ W,
    const float* __restrict__ bias, void* __restrict__ Out_) {
  constexpr int K  = DM;
  constexpr int N  = DM;
  constexpr int BM = 128, BN = 128, BK = 32, LDK = BK + 8;  // pad 8 f16
  __shared__ __align__(16) _Float16 As[BM * LDK];
  __shared__ __align__(16) _Float16 Bs[BN * LDK];

  const int tid    = threadIdx.x;
  const int wave   = tid >> 5;
  const int lane   = tid & 31;
  const int lane16 = lane & 15;
  const int hf     = lane >> 4;       // lane group: 0 = lanes 0-15, 1 = 16-31
  const int wm     = wave >> 2;       // 2 wave rows
  const int wn     = wave & 3;        // 4 wave cols
  const int m0     = blockIdx.y * BM;
  const int n0     = blockIdx.x * BN;

  v8f acc[4][2] = {};

  for (int k0 = 0; k0 < K; k0 += BK) {
    // ---- stage A tile (convert to f16 if needed) ----
    if constexpr (A_F32) {
      const float* A = (const float*)A_;
#pragma unroll
      for (int i = 0; i < 4; ++i) {
        int v = tid + i * 256;            // 1024 float4 = 128x32 fp32
        int r = v >> 3, c = (v & 7) * 4;
        float4 f = *(const float4*)(A + (size_t)(m0 + r) * K + k0 + c);
        v4h t;
        t[0] = (_Float16)f.x; t[1] = (_Float16)f.y;
        t[2] = (_Float16)f.z; t[3] = (_Float16)f.w;
        *(v4h*)&As[r * LDK + c] = t;
      }
    } else {
      const _Float16* A = (const _Float16*)A_;
#pragma unroll
      for (int i = 0; i < 2; ++i) {
        int v = tid + i * 256;            // 512 v8h = 128x32 f16
        int r = v >> 2, c = (v & 3) * 8;
        *(v8h*)&As[r * LDK + c] =
            *(const v8h*)(A + (size_t)(m0 + r) * K + k0 + c);
      }
    }
    // ---- stage W tile (rows = n, cols = k), fp32 -> f16 ----
#pragma unroll
    for (int i = 0; i < 4; ++i) {
      int v = tid + i * 256;
      int r = v >> 3, c = (v & 7) * 4;
      float4 f = *(const float4*)(W + (size_t)(n0 + r) * K + k0 + c);
      v4h t;
      t[0] = (_Float16)f.x; t[1] = (_Float16)f.y;
      t[2] = (_Float16)f.z; t[3] = (_Float16)f.w;
      *(v4h*)&Bs[r * LDK + c] = t;
    }
    __syncthreads();

    // ---- fragments ----
    v16h afrag[4], bfrag[2];
#pragma unroll
    for (int i = 0; i < 4; ++i) {
      // A frag: row M = lane16, elements e -> K = hf*16 + e (contiguous)
      const _Float16* p = &As[(wm * 64 + i * 16 + lane16) * LDK];
      afrag[i] = cat8(*(const v8h*)(p + hf * 16),
                      *(const v8h*)(p + hf * 16 + 8));
    }
#pragma unroll
    for (int j = 0; j < 2; ++j) {
      // B frag: col N = lane16, elements e -> K = hf*16 + e (contiguous)
      const _Float16* p = &Bs[(wn * 32 + j * 16 + lane16) * LDK + hf * 16];
      bfrag[j] = cat8(*(const v8h*)p, *(const v8h*)(p + 8));
    }
#pragma unroll
    for (int i = 0; i < 4; ++i)
#pragma unroll
      for (int j = 0; j < 2; ++j)
        acc[i][j] = wmma_f16(afrag[i], bfrag[j], acc[i][j]);
    __syncthreads();
  }

  // ---- epilogue: bias + store ----
#pragma unroll
  for (int j = 0; j < 2; ++j) {
    const int n  = n0 + wn * 32 + j * 16 + lane16;
    const float bv = bias[n];
#pragma unroll
    for (int i = 0; i < 4; ++i) {
#pragma unroll
      for (int r = 0; r < 8; ++r) {
        const float val = acc[i][j][r] + bv;
        const int m = m0 + wm * 64 + i * 16 + r + hf * 8;
        if constexpr (OUT_HEAD) {
          const int bb = m / SEQ, s = m % SEQ;
          const int hh = n / DK, d = n % DK;
          ((_Float16*)Out_)[(((size_t)bb * NH + hh) * SEQ + s) * DK + d] =
              (_Float16)val;
        } else {
          ((float*)Out_)[(size_t)m * N + n] = val;
        }
      }
    }
  }
}

// ---------------------------------------------------------------------------
// Flash attention: one block = 64 q-rows of one (b, h); 4 waves x 16 rows.
// Iterates over SK in 64-column chunks with online softmax.
// ---------------------------------------------------------------------------
__global__ __launch_bounds__(128) void attn_kernel(
    const _Float16* __restrict__ Qh, const _Float16* __restrict__ Kh,
    const _Float16* __restrict__ Vh, const int* __restrict__ mask,
    _Float16* __restrict__ Xo) {
  constexpr int LDV = DK + 8;  // 72
  __shared__ __align__(16) _Float16 VsT[DK * LDV];    // transposed: [d][sk]
  __shared__ __align__(16) _Float16 Ps[4][16 * LDV];  // per-wave P tile

  const int tid    = threadIdx.x;
  const int wave   = tid >> 5;
  const int lane   = tid & 31;
  const int lane16 = lane & 15;
  const int hf     = lane >> 4;
  const int b      = blockIdx.z;
  const int h      = blockIdx.y;
  const int qbase  = blockIdx.x * 64 + wave * 16;
  const size_t bh  = (size_t)b * NH + h;
  const _Float16* Q  = Qh + bh * SEQ * DK;
  const _Float16* Kp = Kh + bh * SEQ * DK;
  const _Float16* Vp = Vh + bh * SEQ * DK;
  const int* mk = mask + b * SEQ;

  // Q A-fragments for this wave's 16 rows (K dim = 64 -> 2 frags)
  v16h qa[2];
  {
    const _Float16* qr = Q + (size_t)(qbase + lane16) * DK;
#pragma unroll
    for (int f = 0; f < 2; ++f)
      qa[f] = cat8(*(const v8h*)(qr + f * 32 + hf * 16),
                   *(const v8h*)(qr + f * 32 + hf * 16 + 8));
  }

  float mx[8], lsum[8];
#pragma unroll
  for (int r = 0; r < 8; ++r) { mx[r] = -3.0e38f; lsum[r] = 0.0f; }
  v8f acc[4] = {};
  const float scale = 0.125f;  // 1/sqrt(DK)

  for (int sk0 = 0; sk0 < SEQ; sk0 += 64) {
    // ---- stage V chunk transposed into LDS: VsT[d][sk_local] ----
#pragma unroll
    for (int i = 0; i < 4; ++i) {
      int v = tid + i * 128;               // 512 v8h = 64x64 f16
      int r = v >> 3, c = (v & 7) * 8;
      v8h d8 = *(const v8h*)(Vp + (size_t)(sk0 + r) * DK + c);
#pragma unroll
      for (int e = 0; e < 8; ++e) VsT[(c + e) * LDV + r] = d8[e];
    }
    __syncthreads();

    // ---- scores S = scale * Q K^T, mask, chunk max ----
    v8f sc[4];
    float cmax[8];
#pragma unroll
    for (int r = 0; r < 8; ++r) cmax[r] = -3.0e38f;
#pragma unroll
    for (int ct = 0; ct < 4; ++ct) {
      const int skc = sk0 + ct * 16 + lane16;  // this lane's score column
      const _Float16* kr = Kp + (size_t)skc * DK;
      // B frag for K^T: col = sk, row k = d; contiguous in d
      v16h bk0 = cat8(*(const v8h*)(kr + hf * 16),
                      *(const v8h*)(kr + hf * 16 + 8));
      v16h bk1 = cat8(*(const v8h*)(kr + 32 + hf * 16),
                      *(const v8h*)(kr + 32 + hf * 16 + 8));
      v8f s = {};
      s = wmma_f16(qa[0], bk0, s);
      s = wmma_f16(qa[1], bk1, s);
      const bool masked = (mk[skc] == 0);
#pragma unroll
      for (int r = 0; r < 8; ++r) {
        float val = masked ? -1.0e9f : s[r] * scale;
        s[r] = val;
        cmax[r] = fmaxf(cmax[r], val);
      }
      sc[ct] = s;
    }
    // row max across the 16 lanes of each lane group
#pragma unroll
    for (int r = 0; r < 8; ++r)
#pragma unroll
      for (int off = 1; off < 16; off <<= 1)
        cmax[r] = fmaxf(cmax[r], __shfl_xor(cmax[r], off, 32));

    float corr[8], csum[8];
#pragma unroll
    for (int r = 0; r < 8; ++r) {
      const float mn = fmaxf(mx[r], cmax[r]);
      corr[r] = __expf(mx[r] - mn);
      mx[r]   = mn;
      csum[r] = 0.0f;
    }
#pragma unroll
    for (int ct = 0; ct < 4; ++ct)
#pragma unroll
      for (int r = 0; r < 8; ++r) {
        const float p = __expf(sc[ct][r] - mx[r]);
        sc[ct][r] = p;
        csum[r] += p;
      }
#pragma unroll
    for (int r = 0; r < 8; ++r) {
#pragma unroll
      for (int off = 1; off < 16; off <<= 1)
        csum[r] += __shfl_xor(csum[r], off, 32);
      lsum[r] = lsum[r] * corr[r] + csum[r];
    }
#pragma unroll
    for (int dt = 0; dt < 4; ++dt)
#pragma unroll
      for (int r = 0; r < 8; ++r) acc[dt][r] *= corr[r];

    // ---- P: C-layout regs -> LDS -> A-layout frags (wave-private, LDS
    // ops from one wave are in order; no barrier needed) ----
    _Float16* Pw = Ps[wave];
#pragma unroll
    for (int ct = 0; ct < 4; ++ct)
#pragma unroll
      for (int r = 0; r < 8; ++r)
        Pw[(r + hf * 8) * LDV + ct * 16 + lane16] = (_Float16)sc[ct][r];

    v16h pa[2];
    {
      const _Float16* pr = Pw + lane16 * LDV;
#pragma unroll
      for (int f = 0; f < 2; ++f)
        pa[f] = cat8(*(const v8h*)(pr + f * 32 + hf * 16),
                     *(const v8h*)(pr + f * 32 + hf * 16 + 8));
    }

    // ---- acc += P @ V ----
#pragma unroll
    for (int dt = 0; dt < 4; ++dt) {
      const _Float16* vr = &VsT[(dt * 16 + lane16) * LDV];
      v16h vb0 = cat8(*(const v8h*)(vr + hf * 16),
                      *(const v8h*)(vr + hf * 16 + 8));
      v16h vb1 = cat8(*(const v8h*)(vr + 32 + hf * 16),
                      *(const v8h*)(vr + 32 + hf * 16 + 8));
      acc[dt] = wmma_f16(pa[0], vb0, acc[dt]);
      acc[dt] = wmma_f16(pa[1], vb1, acc[dt]);
    }
    __syncthreads();
  }

  // ---- normalize and store x[b, s, h*64+d] as f16 ----
#pragma unroll
  for (int dt = 0; dt < 4; ++dt) {
    const int d = dt * 16 + lane16;
#pragma unroll
    for (int r = 0; r < 8; ++r) {
      const int qrow = qbase + r + hf * 8;
      const float o = acc[dt][r] / lsum[r];
      Xo[((size_t)b * SEQ + qrow) * DM + h * DK + d] = (_Float16)o;
    }
  }
}

// ---------------------------------------------------------------------------
extern "C" void kernel_launch(void* const* d_in, const int* in_sizes, int n_in,
                              void* d_out, int out_size, void* d_ws,
                              size_t ws_size, hipStream_t stream) {
  (void)in_sizes; (void)n_in; (void)out_size; (void)ws_size;
  const float* q   = (const float*)d_in[0];
  const float* k   = (const float*)d_in[1];
  const float* v   = (const float*)d_in[2];
  const int*   msk = (const int*)d_in[3];
  const float* wq  = (const float*)d_in[4];
  const float* bq  = (const float*)d_in[5];
  const float* wk  = (const float*)d_in[6];
  const float* bk  = (const float*)d_in[7];
  const float* wv  = (const float*)d_in[8];
  const float* bv  = (const float*)d_in[9];
  const float* wo  = (const float*)d_in[10];
  const float* bo  = (const float*)d_in[11];

  const size_t perT = (size_t)BSZ * NH * SEQ * DK;  // f16 elements per tensor
  _Float16* qh = (_Float16*)d_ws;
  _Float16* kh = qh + perT;
  _Float16* vh = kh + perT;
  _Float16* xf = vh + perT;  // [B, S, DM] f16

  const dim3 gg(DM / 128, (BSZ * SEQ) / 128);  // (6, 32)
  gemm_wt_kernel<true, true><<<gg, 256, 0, stream>>>(q, wq, bq, qh);
  gemm_wt_kernel<true, true><<<gg, 256, 0, stream>>>(k, wk, bk, kh);
  gemm_wt_kernel<true, true><<<gg, 256, 0, stream>>>(v, wv, bv, vh);
  attn_kernel<<<dim3(SEQ / 64, NH, BSZ), 128, 0, stream>>>(qh, kh, vh, msk, xf);
  gemm_wt_kernel<false, false><<<gg, 256, 0, stream>>>(xf, wo, bo, d_out);
}
